// GATLayer_10797547782501
// MI455X (gfx1250) — matrix operand order repaired and checked by reading, not compiled, not run on hardware
//
#include <hip/hip_runtime.h>
#include <hip/hip_bf16.h>
#include <stdint.h>

// RGATConv (heads=1) for MI455X / gfx1250, wave32.
//   hx = x @ W_r            -> v_wmma_f32_16x16x32_bf16 (f32 accum)
//   nq = x @ (W_r q), nk = x @ (W_r k)   (algebraic refactor, avoids re-reading hx)
//   logits -> segment softmax via ordered-int atomicMax + atomicAdd
//   agg[dst] += alpha * hx[r,src]  (float4 gather + f32 atomics, L2-resident)
// Workspace use: ~187 MB (hx dominates at R*N*D*4 = 153.6 MB; fits the 192MB L2).

typedef __attribute__((ext_vector_type(16))) __bf16 v16bf;
typedef __attribute__((ext_vector_type(8)))  float  v8f;

#define DD 128
#define NEG_SLOPE 0.2f

// ---------------------------------------------------------------- small prep
// Wt[r][o][d] = bf16(W[r][d][o])  (transpose so B-fragments load contiguously)
__global__ void rgat_wt(const float* __restrict__ W, __bf16* __restrict__ Wt, int RDD) {
    int idx = blockIdx.x * blockDim.x + threadIdx.x;
    if (idx >= RDD) return;
    int r = idx / (DD * DD);
    int rem = idx - r * DD * DD;
    int o = rem / DD, d = rem - o * DD;
    Wt[idx] = (__bf16)W[((size_t)r * DD + d) * DD + o];
}

// wq[r*D+d] = sum_o W[r,d,o] * q[o] ; same for k
__global__ void rgat_wqk(const float* __restrict__ W, const float* __restrict__ q,
                         const float* __restrict__ k, float* __restrict__ wq,
                         float* __restrict__ wk, int RD) {
    int idx = blockIdx.x * blockDim.x + threadIdx.x;
    if (idx >= RD) return;
    const float* wrow = W + (size_t)idx * DD;
    float sq = 0.f, sk = 0.f;
    for (int o = 0; o < DD; ++o) { float w = wrow[o]; sq += w * q[o]; sk += w * k[o]; }
    wq[idx] = sq; wk[idx] = sk;
}

// ---------------------------------------------------------------- WMMA GEMM
// Wave-per-strip: each wave computes 16 rows x 128 cols of hx[r] with
// 32x v_wmma_f32_16x16x32_bf16 (8 N-tiles * 4 K-steps).
__global__ void __launch_bounds__(256) rgat_gemm(const float* __restrict__ x,
                                                 const __bf16* __restrict__ Wt,
                                                 float* __restrict__ hx, int N) {
    const int r    = blockIdx.y;
    const int wave = threadIdx.x >> 5;
    const int lane = threadIdx.x & 31;
    const int strip = blockIdx.x * 8 + wave;
    if (strip * 16 >= N) return;                    // wave-uniform: EXEC stays all-1s
    const int nb    = strip * 16;
    const int laneM = lane & 15;
    const int hiA   = (lane >> 4) ? 16 : 0;         // A-frag K split (contiguous k = 16h + s, same as B)
    const int hiB   = (lane >> 4) ? 16 : 0;         // B-frag K split

    v8f acc[8] = {};
    const float*  xrow  = x  + (size_t)(nb + laneM) * DD;
    const __bf16* wbase = Wt + (size_t)r * DD * DD;

    #pragma unroll
    for (int ks = 0; ks < 4; ++ks) {
        const int kA = ks * 32;
        const float* xp = xrow + kA + hiA;
        float4 a0 = *(const float4*)(xp +  0);
        float4 a1 = *(const float4*)(xp +  4);
        float4 a2 = *(const float4*)(xp +  8);
        float4 a3 = *(const float4*)(xp + 12);
        v16bf av;
        av[ 0]=(__bf16)a0.x; av[ 1]=(__bf16)a0.y; av[ 2]=(__bf16)a0.z; av[ 3]=(__bf16)a0.w;
        av[ 4]=(__bf16)a1.x; av[ 5]=(__bf16)a1.y; av[ 6]=(__bf16)a1.z; av[ 7]=(__bf16)a1.w;
        av[ 8]=(__bf16)a2.x; av[ 9]=(__bf16)a2.y; av[10]=(__bf16)a2.z; av[11]=(__bf16)a2.w;
        av[12]=(__bf16)a3.x; av[13]=(__bf16)a3.y; av[14]=(__bf16)a3.z; av[15]=(__bf16)a3.w;
        #pragma unroll
        for (int t = 0; t < 8; ++t) {
            const __bf16* bp = wbase + (size_t)(t * 16 + laneM) * DD + kA + hiB;
            v16bf bv = *(const v16bf*)bp;           // 16 contiguous bf16 (32B aligned)
            acc[t] = __builtin_amdgcn_wmma_f32_16x16x32_bf16(
                         false, av, false, bv, (short)0, acc[t], false, false);
        }
    }
    const int mHalf = (lane >> 4) ? 8 : 0;
    #pragma unroll
    for (int t = 0; t < 8; ++t)
        #pragma unroll
        for (int v = 0; v < 8; ++v) {
            int row = nb + v + mHalf;               // C/D layout: VGPR v -> row v(+8)
            hx[((size_t)r * N + row) * DD + t * 16 + laneM] = acc[t][v];
        }
}

// ---------------------------------------------------------------- nq / nk
// One wave per node per relation: lane covers 4 channels, butterfly reduce.
__global__ void rgat_nqnk(const float* __restrict__ x, const float* __restrict__ wq,
                          const float* __restrict__ wk, float* __restrict__ nq,
                          float* __restrict__ nk, int N) {
    const int r = blockIdx.y;
    const int wave = threadIdx.x >> 5, lane = threadIdx.x & 31;
    const int n = blockIdx.x * 8 + wave;
    if (n >= N) return;
    float4 xv = ((const float4*)(x  + (size_t)n * DD))[lane];
    float4 qv = ((const float4*)(wq + r * DD))[lane];
    float4 kv = ((const float4*)(wk + r * DD))[lane];
    float sq = xv.x*qv.x + xv.y*qv.y + xv.z*qv.z + xv.w*qv.w;
    float sk = xv.x*kv.x + xv.y*kv.y + xv.z*kv.z + xv.w*kv.w;
    for (int off = 16; off; off >>= 1) {
        sq += __shfl_xor(sq, off, 32);
        sk += __shfl_xor(sk, off, 32);
    }
    if (lane == 0) { nq[(size_t)r * N + n] = sq; nk[(size_t)r * N + n] = sk; }
}

// ---------------------------------------------------------------- init (every launch)
__global__ void rgat_init(int* __restrict__ mEnc, float* __restrict__ ssum,
                          float* __restrict__ agg, int N, int ND) {
    int idx = blockIdx.x * blockDim.x + threadIdx.x;
    if (idx < ND) agg[idx] = 0.f;
    if (idx < N) { mEnc[idx] = (int)0x80000000; ssum[idx] = 0.f; }
}

__device__ __forceinline__ int fenc(float f) {
    int i = __float_as_int(f);
    return i >= 0 ? i : (i ^ 0x7fffffff);           // monotone float<->int bijection
}
__device__ __forceinline__ float fdec(int i) {
    return __int_as_float(i >= 0 ? i : (i ^ 0x7fffffff));
}

// logits + segment max
__global__ void rgat_logits(const int* __restrict__ ei, const int* __restrict__ et,
                            const float* __restrict__ nq, const float* __restrict__ nk,
                            float* __restrict__ logits, int* __restrict__ mEnc,
                            int E, int N) {
    int e = blockIdx.x * blockDim.x + threadIdx.x;
    if (e >= E) return;
    int s = ei[e], d = ei[E + e], r = et[e];
    float lg = nq[(size_t)r * N + d] + nk[(size_t)r * N + s];
    lg = lg > 0.f ? lg : NEG_SLOPE * lg;
    logits[e] = lg;
    atomicMax(mEnc + d, fenc(lg));
}

// exp(l - m[dst]) + segment sum  (logits buffer becomes exp buffer in place)
__global__ void rgat_exp(const int* __restrict__ ei, float* __restrict__ logits,
                         const int* __restrict__ mEnc, float* __restrict__ ssum, int E) {
    int e = blockIdx.x * blockDim.x + threadIdx.x;
    if (e >= E) return;
    int d = ei[E + e];
    float ex = expf(logits[e] - fdec(mEnc[d]));
    logits[e] = ex;
    atomicAdd(ssum + d, ex);
}

// alpha + message scatter: one wave per edge, lane handles 4 channels
__global__ void __launch_bounds__(256) rgat_scatter(const int* __restrict__ ei,
        const int* __restrict__ et, const float* __restrict__ expv,
        const float* __restrict__ ssum, const float* __restrict__ hx,
        float* __restrict__ agg, float* __restrict__ outAlpha, int E, int N) {
    const int wave = threadIdx.x >> 5, lane = threadIdx.x & 31;
    const int e = blockIdx.x * 8 + wave;
    if (e >= E) return;
    int s = ei[e], d = ei[E + e], r = et[e];
    float alpha = expv[e] / (ssum[d] + 1e-16f);
    if (lane == 0) outAlpha[e] = alpha;
    float4 h = ((const float4*)(hx + ((size_t)r * N + s) * DD))[lane];
    float* ap = agg + (size_t)d * DD + lane * 4;
    atomicAdd(ap + 0, alpha * h.x);
    atomicAdd(ap + 1, alpha * h.y);
    atomicAdd(ap + 2, alpha * h.z);
    atomicAdd(ap + 3, alpha * h.w);
}

// out = agg + bias + x (residual)
__global__ void rgat_final(const float* __restrict__ agg, const float* __restrict__ bias,
                           const float* __restrict__ x, float* __restrict__ out, int ND) {
    int idx = blockIdx.x * blockDim.x + threadIdx.x;
    if (idx >= ND) return;
    out[idx] = agg[idx] + bias[idx & (DD - 1)] + x[idx];
}

// ---------------------------------------------------------------- launch
extern "C" void kernel_launch(void* const* d_in, const int* in_sizes, int n_in,
                              void* d_out, int out_size, void* d_ws, size_t ws_size,
                              hipStream_t stream) {
    const float* x    = (const float*)d_in[0];
    const int*   ei   = (const int*)  d_in[1];
    const int*   et   = (const int*)  d_in[2];
    const float* W    = (const float*)d_in[3];
    const float* q    = (const float*)d_in[4];
    const float* k    = (const float*)d_in[5];
    const float* bias = (const float*)d_in[6];

    const int N = in_sizes[0] / DD;          // 50000 (multiple of 16)
    const int E = in_sizes[2];               // 1200000
    const int R = in_sizes[3] / (DD * DD);   // 6
    const int ND = N * DD;

    // workspace carve (256B aligned)
    char* p = (char*)d_ws;
    auto carve = [&](size_t bytes) -> void* {
        void* r = (void*)p; p += (bytes + 255) & ~(size_t)255; return r;
    };
    float*  hx     = (float*) carve((size_t)R * ND * sizeof(float));   // 153.6 MB
    __bf16* Wt     = (__bf16*)carve((size_t)R * DD * DD * sizeof(__bf16));
    float*  wq     = (float*) carve((size_t)R * DD * sizeof(float));
    float*  wk     = (float*) carve((size_t)R * DD * sizeof(float));
    float*  nq     = (float*) carve((size_t)R * N * sizeof(float));
    float*  nk     = (float*) carve((size_t)R * N * sizeof(float));
    float*  logits = (float*) carve((size_t)E * sizeof(float));        // reused as exp
    int*    mEnc   = (int*)   carve((size_t)N * sizeof(int));
    float*  ssum   = (float*) carve((size_t)N * sizeof(float));
    float*  agg    = (float*) carve((size_t)ND * sizeof(float));

    float* outFeat  = (float*)d_out;         // [N, D]
    float* outAlpha = outFeat + (size_t)ND;  // [E, 1]

    const int B = 256;
    // prep
    rgat_wt <<<(R * DD * DD + B - 1) / B, B, 0, stream>>>(W, Wt, R * DD * DD);
    rgat_wqk<<<(R * DD + B - 1) / B,      B, 0, stream>>>(W, q, k, wq, wk, R * DD);
    rgat_init<<<(ND + B - 1) / B,         B, 0, stream>>>(mEnc, ssum, agg, N, ND);
    // hx = x @ W_r  (WMMA)
    dim3 gGemm(((N + 15) / 16 + 7) / 8, R);
    rgat_gemm<<<gGemm, B, 0, stream>>>(x, Wt, hx, N);
    // nq/nk = x @ wq/wk
    dim3 gNqk((N + 7) / 8, R);
    rgat_nqnk<<<gNqk, B, 0, stream>>>(x, wq, wk, nq, nk, N);
    // edge phase
    rgat_logits<<<(E + B - 1) / B, B, 0, stream>>>(ei, et, nq, nk, logits, mEnc, E, N);
    rgat_exp   <<<(E + B - 1) / B, B, 0, stream>>>(ei, logits, mEnc, ssum, E);
    rgat_scatter<<<(E + 7) / 8,    B, 0, stream>>>(ei, et, logits, ssum, hx, agg,
                                                   outAlpha, E, N);
    rgat_final <<<(ND + B - 1) / B, B, 0, stream>>>(agg, bias, x, outFeat, ND);
}
